// DGCNN_77867757076593
// MI455X (gfx1250) — matrix operand order repaired and checked by reading, not compiled, not run on hardware
//
#include <hip/hip_runtime.h>
#include <hip/hip_bf16.h>

typedef __attribute__((ext_vector_type(16))) _Float16 v16h;
typedef __attribute__((ext_vector_type(4)))  _Float16 v4h;
typedef __attribute__((ext_vector_type(8)))  float    v8f;
typedef __attribute__((ext_vector_type(4)))  float    v4f;

#define NPTS 2048
#define BATCH 4

// Permute K index so each lane's A fragment (K: {kh..kh+7} U {16+kh..16+kh+7})
// is contiguous in LDS: c -> (c&7) | ((c&8)<<1) | ((c&16)>>1)
__device__ __forceinline__ int kperm(int c) {
    return (c & 7) | ((c & 8) << 1) | ((c & 16) >> 1);
}

// ---------------------------------------------------------------------------
// Tiled f16 WMMA GEMM: out[b, o, n] = sum_c weight(o,c) * X[b, c, n]
//  MODE 0: weight(o,c) = Wg[o*ldw + woff + c]                    (plain / W_lo)
//  MODE 1: weight(o,c) = Wg[o*ldw + woff + c] - Wg[o*ldw + c]    (W_hi - W_lo)
//  MODE 2: weight(o,c) = Wg[b*strideW + c*N + o]                 (X^T, Gram/KNN)
// Workgroup tile 64(M) x 128(N) x 32(K); 8 waves; each wave owns a 32x32
// output block = 2x2 grid of 16x16 WMMA tiles (4 v_wmma per K step).
// LDS tiles stored in fragment order: every lane fragment = contiguous 32B.
// Staging is fully unrolled with pointer-increment addressing (no per-lane
// trip counts, no per-element 64-bit multiplies).
// ---------------------------------------------------------------------------
template <int MODE>
__global__ __launch_bounds__(256)
void k_gemm(const float* __restrict__ Wg, int ldw, int woff,
            const float* __restrict__ Xg, long strideX,
            float* __restrict__ out, int M, int N, int K, long strideW)
{
    __shared__ _Float16 Al[64][32];    // [row][perm(k)]  (4 KB)
    __shared__ _Float16 Bl[128][32];   // [col][perm(k)]  (8 KB)

    const int tid  = threadIdx.x;
    const int lane = tid & 31;
    const int wid  = tid >> 5;
    const int wm   = wid & 1;          // 2 waves across M
    const int wn   = wid >> 1;         // 4 waves across N
    const int b     = blockIdx.z;
    const int oBase = blockIdx.y * 64;
    const int nBase = blockIdx.x * 128;

    const int half = lane >> 4;        // 0/1: selects fragment half
    const int lc   = lane & 15;
    const int koff = half * 16;

    v8f acc00 = {}, acc01 = {}, acc10 = {}, acc11 = {};

    // ---- per-thread staging geometry (constant across K loop) ----
    // A (MODE 0/1): 2 iters, each a float4 along c
    const int aRow = tid >> 3;             // 0..31  (+32 on iter 1)
    const int aCol = (tid & 7) * 4;        // 0..28
    const int aDst = kperm(aCol);          // 4-run stays contiguous
    // A (MODE 2): 2 iters, each a float4 along o
    const int tRow = (tid & 15) * 4;       // o0: 0..60
    const int tCol = tid >> 4;             // c : 0..15 (+16 on iter 1)
    const int tDst = kperm(tCol);          // iter1 dest = tDst + 8
    // B: 4 iters, each a float4 along n
    const int bCol = (tid & 31) * 4;       // n0: 0..124
    const int bRow = tid >> 5;             // c : 0..7 (+8 per iter)

    const float* aPtr   = Wg + (size_t)(oBase + aRow) * ldw + woff + aCol;
    const float* aPtrLo = Wg + (size_t)(oBase + aRow) * ldw + aCol;          // MODE 1
    const float* tPtr   = Wg + (size_t)b * strideW + (size_t)tCol * N + oBase + tRow; // MODE 2
    const float* bPtr   = Xg + (size_t)b * strideX + (size_t)bRow * N + nBase + bCol;
    const size_t aRowLdw = (size_t)32 * ldw;

    const int kmain = K & ~31;
    for (int kk = 0; kk < kmain; kk += 32) {
        // ---------------- branch-free staging ----------------
        if (MODE == 2) {
            #pragma unroll
            for (int it = 0; it < 2; ++it) {
                const v4f w = *(const v4f*)(tPtr + (size_t)(16 * it) * N);
                const int pc = tDst + 8 * it;
                const int o0 = tRow;
                Al[o0 + 0][pc] = (_Float16)w.x;
                Al[o0 + 1][pc] = (_Float16)w.y;
                Al[o0 + 2][pc] = (_Float16)w.z;
                Al[o0 + 3][pc] = (_Float16)w.w;
            }
            tPtr += (size_t)32 * N;
        } else {
            #pragma unroll
            for (int it = 0; it < 2; ++it) {
                v4f w = *(const v4f*)(aPtr + aRowLdw * it);
                if (MODE == 1) {
                    const v4f wl = *(const v4f*)(aPtrLo + aRowLdw * it);
                    w.x -= wl.x; w.y -= wl.y; w.z -= wl.z; w.w -= wl.w;
                }
                v4h hv = { (_Float16)w.x, (_Float16)w.y, (_Float16)w.z, (_Float16)w.w };
                *(v4h*)&Al[aRow + 32 * it][aDst] = hv;
            }
            aPtr += 32; aPtrLo += 32;
        }
        #pragma unroll
        for (int it = 0; it < 4; ++it) {
            const v4f x = *(const v4f*)(bPtr + (size_t)(8 * it) * N);
            const int c = kperm(bRow + 8 * it);
            Bl[bCol + 0][c] = (_Float16)x.x;
            Bl[bCol + 1][c] = (_Float16)x.y;
            Bl[bCol + 2][c] = (_Float16)x.z;
            Bl[bCol + 3][c] = (_Float16)x.w;
        }
        bPtr += (size_t)32 * N;
        __syncthreads();

        // ---- fragments: contiguous 32B per lane -> 2x ds_load_b128 each ----
        const v16h a0 = *(const v16h*)&Al[wm * 32 + lc][koff];
        const v16h a1 = *(const v16h*)&Al[wm * 32 + 16 + lc][koff];
        const v16h b0 = *(const v16h*)&Bl[wn * 32 + lc][koff];
        const v16h b1 = *(const v16h*)&Bl[wn * 32 + 16 + lc][koff];

        acc00 = __builtin_amdgcn_wmma_f32_16x16x32_f16(false, a0, false, b0, (short)0, acc00, false, false);
        acc01 = __builtin_amdgcn_wmma_f32_16x16x32_f16(false, a0, false, b1, (short)0, acc01, false, false);
        acc10 = __builtin_amdgcn_wmma_f32_16x16x32_f16(false, a1, false, b0, (short)0, acc10, false, false);
        acc11 = __builtin_amdgcn_wmma_f32_16x16x32_f16(false, a1, false, b1, (short)0, acc11, false, false);
        __syncthreads();
    }

    if (kmain < K) {
        // ---------------- guarded remainder tile (only K=17 block 1) --------
        for (int e = tid; e < 64 * 32; e += 256) {
            int o = e >> 5, c = e & 31;
            int kg = kmain + c;
            float val = 0.f;
            if (kg < K) {
                int og = oBase + o;
                if (MODE == 0)
                    val = Wg[(size_t)og * ldw + woff + kg];
                else if (MODE == 1)
                    val = Wg[(size_t)og * ldw + woff + kg] - Wg[(size_t)og * ldw + kg];
                else
                    val = Wg[(size_t)b * strideW + (size_t)kg * N + og];
            }
            Al[o][kperm(c)] = (_Float16)val;
        }
        for (int e = tid; e < 128 * 32; e += 256) {
            int n = e >> 5, c = e & 31;
            int kg = kmain + c;
            float val = 0.f;
            if (kg < K)
                val = Xg[(size_t)b * strideX + (size_t)kg * N + nBase + n];
            Bl[n][kperm(c)] = (_Float16)val;
        }
        __syncthreads();

        const v16h a0 = *(const v16h*)&Al[wm * 32 + lc][koff];
        const v16h a1 = *(const v16h*)&Al[wm * 32 + 16 + lc][koff];
        const v16h b0 = *(const v16h*)&Bl[wn * 32 + lc][koff];
        const v16h b1 = *(const v16h*)&Bl[wn * 32 + 16 + lc][koff];

        acc00 = __builtin_amdgcn_wmma_f32_16x16x32_f16(false, a0, false, b0, (short)0, acc00, false, false);
        acc01 = __builtin_amdgcn_wmma_f32_16x16x32_f16(false, a0, false, b1, (short)0, acc01, false, false);
        acc10 = __builtin_amdgcn_wmma_f32_16x16x32_f16(false, a1, false, b0, (short)0, acc10, false, false);
        acc11 = __builtin_amdgcn_wmma_f32_16x16x32_f16(false, a1, false, b1, (short)0, acc11, false, false);
        __syncthreads();
    }

    // ---- store D tiles (8-VGPR f32 C/D layout) ----
    const int r0 = oBase + wm * 32 + 8 * half;
    const int c0 = nBase + wn * 32 + lc;
    float* op = out + ((size_t)b * M + r0) * N + c0;
    #pragma unroll
    for (int r = 0; r < 8; ++r) {
        op[(size_t)r * N]             = acc00[r];
        op[(size_t)r * N + 16]        = acc01[r];
        op[(size_t)(r + 16) * N]      = acc10[r];
        op[(size_t)(r + 16) * N + 16] = acc11[r];
    }
}

// ---------------------------------------------------------------------------
// Split inputs: x (B,17,N) and idx0 (B,N,3)
// ---------------------------------------------------------------------------
__global__ void k_prep(const float* __restrict__ in, float* __restrict__ xb,
                       int* __restrict__ idx0)
{
    int gid = blockIdx.x * 256 + threadIdx.x;
    const int tot = BATCH * 20 * NPTS;
    if (gid >= tot) return;
    int n = gid % NPTS;
    int rest = gid / NPTS;
    int c = rest % 20, b = rest / 20;
    float v = in[gid];
    if (c < 17) xb[((size_t)b * 17 + c) * NPTS + n] = v;
    else        idx0[((size_t)b * NPTS + n) * 3 + (c - 17)] = (int)v;
}

__global__ void k_zero(float* p, int n)
{
    int i = blockIdx.x * 256 + threadIdx.x;
    if (i < n) p[i] = 0.f;
}

// ---------------------------------------------------------------------------
// Edge-block stats: y = v[b,o,n] + u[b,o,idx]; accumulate sum/sumsq per channel
// ---------------------------------------------------------------------------
__global__ void k_edge_stats(const float* __restrict__ u, const float* __restrict__ v,
                             const int* __restrict__ idx, int k, int O, int N,
                             int rows, float* __restrict__ sum, float* __restrict__ sumsq)
{
    const int o = threadIdx.x;
    long row0 = (long)blockIdx.x * rows;
    float s1 = 0.f, s2 = 0.f;
    for (int r = 0; r < rows; ++r) {
        long bn = row0 + r;
        int b = (int)(bn / N), n = (int)(bn % N);
        const float* ub = u + ((size_t)b * O + o) * N;
        float vv = v[((size_t)b * O + o) * N + n];
        const int* ip = idx + ((size_t)b * N + n) * k;
        for (int j = 0; j < k; ++j) {
            float y = vv + ub[ip[j]];
            s1 += y; s2 += y * y;
        }
    }
    atomicAdd(&sum[o], s1);
    atomicAdd(&sumsq[o], s2);
}

__global__ void k_finalize(const float* __restrict__ sum, const float* __restrict__ sumsq,
                           const float* __restrict__ g, const float* __restrict__ bb,
                           float cnt, float* __restrict__ s, float* __restrict__ t, int O)
{
    int o = blockIdx.x * blockDim.x + threadIdx.x;
    if (o >= O) return;
    float m   = sum[o] / cnt;
    float var = sumsq[o] / cnt - m * m;
    float sc  = g[o] * rsqrtf(var + 1e-5f);
    s[o] = sc;
    t[o] = bb[o] - m * sc;
}

// ---------------------------------------------------------------------------
// Edge-block output: max_j lrelu(BN(v + u[idx_j])) -> xc slab
// ---------------------------------------------------------------------------
__global__ void k_edge_out(const float* __restrict__ u, const float* __restrict__ v,
                           const int* __restrict__ idx, int k, int O, int N, int rows,
                           const float* __restrict__ s, const float* __restrict__ t,
                           float* __restrict__ outp, long outStride)
{
    const int o = threadIdx.x;
    long row0 = (long)blockIdx.x * rows;
    const float sc = s[o], sh = t[o];
    for (int r = 0; r < rows; ++r) {
        long bn = row0 + r;
        int b = (int)(bn / N), n = (int)(bn % N);
        const float* ub = u + ((size_t)b * O + o) * N;
        float vv = v[((size_t)b * O + o) * N + n];
        const int* ip = idx + ((size_t)b * N + n) * k;
        float mx = -3.4e38f;
        for (int j = 0; j < k; ++j) {
            float y = (vv + ub[ip[j]]) * sc + sh;
            y = (y >= 0.f) ? y : 0.2f * y;
            mx = fmaxf(mx, y);
        }
        outp[(size_t)b * outStride + (size_t)o * N + n] = mx;
    }
}

// ---------------------------------------------------------------------------
// KNN: squared norms, then top-20 of d = 2*inner - xx_n - xx_m per row
// ---------------------------------------------------------------------------
__global__ void k_xx(const float* __restrict__ X, long strideX, int C, int N,
                     float* __restrict__ xx)
{
    long gid = (long)blockIdx.x * 256 + threadIdx.x;
    if (gid >= (long)BATCH * N) return;
    int b = (int)(gid / N), n = (int)(gid % N);
    float s = 0.f;
    for (int c = 0; c < C; ++c) {
        float v = X[(size_t)b * strideX + (size_t)c * N + n];
        s += v * v;
    }
    xx[gid] = s;
}

__global__ void k_topk(const float* __restrict__ inner, const float* __restrict__ xx,
                       int* __restrict__ kidx, int N)
{
    long gid = (long)blockIdx.x * 256 + threadIdx.x;
    if (gid >= (long)BATCH * N) return;
    int b = (int)(gid / N), n = (int)(gid % N);
    const float* row = inner + ((size_t)b * N + n) * N;
    const float* xxb = xx + (size_t)b * N;
    const float xn = xxb[n];
    float bd[20]; int bi[20];
    #pragma unroll
    for (int j = 0; j < 20; ++j) { bd[j] = -3.4e38f; bi[j] = 0; }
    for (int m = 0; m < N; ++m) {
        float dv = 2.f * row[m] - xn - xxb[m];
        if (dv > bd[19]) {
            int p = 19;
            while (p > 0 && dv > bd[p - 1]) { bd[p] = bd[p - 1]; bi[p] = bi[p - 1]; --p; }
            bd[p] = dv; bi[p] = m;
        }
    }
    int* op = kidx + gid * 20;
    #pragma unroll
    for (int j = 0; j < 20; ++j) op[j] = bi[j];
}

// ---------------------------------------------------------------------------
// Conv7 stats (per-channel mean/var over B,N) and BN + lrelu + max/mean pool
// ---------------------------------------------------------------------------
__global__ void k_conv7_stats(const float* __restrict__ y, int N,
                              float* __restrict__ sum, float* __restrict__ sumsq)
{
    __shared__ float s1[256], s2[256];
    const int o = blockIdx.x, tid = threadIdx.x;
    float a1 = 0.f, a2 = 0.f;
    const int tot = BATCH * N;
    for (int i = tid; i < tot; i += 256) {
        int b = i / N, n = i % N;
        float val = y[((size_t)b * 1024 + o) * N + n];
        a1 += val; a2 += val * val;
    }
    s1[tid] = a1; s2[tid] = a2; __syncthreads();
    for (int st = 128; st > 0; st >>= 1) {
        if (tid < st) { s1[tid] += s1[tid + st]; s2[tid] += s2[tid + st]; }
        __syncthreads();
    }
    if (tid == 0) { sum[o] = s1[0]; sumsq[o] = s2[0]; }
}

__global__ void k_conv7_out(const float* __restrict__ y, int N,
                            const float* __restrict__ s, const float* __restrict__ t,
                            float* __restrict__ h)
{
    __shared__ float sm[256], ss[256];
    const int bo = blockIdx.x;
    const int b = bo >> 10, o = bo & 1023;
    const int tid = threadIdx.x;
    const float* yb = y + ((size_t)b * 1024 + o) * N;
    const float sc = s[o], sh = t[o];
    float mx = -3.4e38f, sum = 0.f;
    for (int n = tid; n < N; n += 256) {
        float val = yb[n] * sc + sh;
        val = (val >= 0.f) ? val : 0.2f * val;
        mx = fmaxf(mx, val); sum += val;
    }
    sm[tid] = mx; ss[tid] = sum; __syncthreads();
    for (int st = 128; st > 0; st >>= 1) {
        if (tid < st) { sm[tid] = fmaxf(sm[tid], sm[tid + st]); ss[tid] += ss[tid + st]; }
        __syncthreads();
    }
    if (tid == 0) {
        h[(size_t)b * 2048 + o]        = sm[0];
        h[(size_t)b * 2048 + 1024 + o] = ss[0] / (float)N;
    }
}

// ---------------------------------------------------------------------------
// MLP head: (4,2048) -> 512 -> 256 -> 64 -> 3, BN over batch axis; one WG.
// ---------------------------------------------------------------------------
__global__ __launch_bounds__(256)
void k_mlp(const float* __restrict__ h,
           const float* __restrict__ L1, const float* __restrict__ g8,  const float* __restrict__ b8,
           const float* __restrict__ L2, const float* __restrict__ bl2, const float* __restrict__ g9,  const float* __restrict__ b9,
           const float* __restrict__ L3, const float* __restrict__ bl3, const float* __restrict__ g10, const float* __restrict__ b10,
           const float* __restrict__ L4, const float* __restrict__ bl4,
           float* __restrict__ outp)
{
    __shared__ float hs[4 * 2048];
    __shared__ float t1[4 * 512];
    __shared__ float t2[4 * 256];
    __shared__ float t3[4 * 64];
    const int tid = threadIdx.x;

    for (int i = tid; i < 4 * 2048; i += 256) hs[i] = h[i];
    __syncthreads();

    for (int i = tid; i < 4 * 512; i += 256) {
        int r = i >> 9, o = i & 511;
        const float* w = L1 + (size_t)o * 2048;
        float acc = 0.f;
        for (int c = 0; c < 2048; ++c) acc += hs[r * 2048 + c] * w[c];
        t1[i] = acc;
    }
    __syncthreads();
    for (int o = tid; o < 512; o += 256) {
        float m = 0.f;
        for (int r = 0; r < 4; ++r) m += t1[r * 512 + o];
        m *= 0.25f;
        float var = 0.f;
        for (int r = 0; r < 4; ++r) { float d = t1[r * 512 + o] - m; var += d * d; }
        var *= 0.25f;
        float sc = g8[o] * rsqrtf(var + 1e-5f), sh = b8[o] - m * sc;
        for (int r = 0; r < 4; ++r) {
            float y = t1[r * 512 + o] * sc + sh;
            t1[r * 512 + o] = (y >= 0.f) ? y : 0.2f * y;
        }
    }
    __syncthreads();

    for (int i = tid; i < 4 * 256; i += 256) {
        int r = i >> 8, o = i & 255;
        const float* w = L2 + (size_t)o * 512;
        float acc = bl2[o];
        for (int c = 0; c < 512; ++c) acc += t1[r * 512 + c] * w[c];
        t2[i] = acc;
    }
    __syncthreads();
    if (tid < 256) {
        int o = tid;
        float m = 0.f;
        for (int r = 0; r < 4; ++r) m += t2[r * 256 + o];
        m *= 0.25f;
        float var = 0.f;
        for (int r = 0; r < 4; ++r) { float d = t2[r * 256 + o] - m; var += d * d; }
        var *= 0.25f;
        float sc = g9[o] * rsqrtf(var + 1e-5f), sh = b9[o] - m * sc;
        for (int r = 0; r < 4; ++r) {
            float y = t2[r * 256 + o] * sc + sh;
            t2[r * 256 + o] = (y >= 0.f) ? y : 0.2f * y;
        }
    }
    __syncthreads();

    if (tid < 256) {
        int r = tid >> 6, o = tid & 63;
        const float* w = L3 + (size_t)o * 256;
        float acc = bl3[o];
        for (int c = 0; c < 256; ++c) acc += t2[r * 256 + c] * w[c];
        t3[tid] = acc;
    }
    __syncthreads();
    if (tid < 64) {
        int o = tid;
        float m = 0.f;
        for (int r = 0; r < 4; ++r) m += t3[r * 64 + o];
        m *= 0.25f;
        float var = 0.f;
        for (int r = 0; r < 4; ++r) { float d = t3[r * 64 + o] - m; var += d * d; }
        var *= 0.25f;
        float sc = g10[o] * rsqrtf(var + 1e-5f), sh = b10[o] - m * sc;
        for (int r = 0; r < 4; ++r) {
            float y = t3[r * 64 + o] * sc + sh;
            t3[r * 64 + o] = (y >= 0.f) ? y : 0.2f * y;
        }
    }
    __syncthreads();

    if (tid < 12) {
        int r = tid / 3, o = tid % 3;
        float acc = bl4[o];
        const float* w = L4 + (size_t)o * 64;
        for (int c = 0; c < 64; ++c) acc += t3[r * 64 + c] * w[c];
        outp[tid] = acc;
    }
}

// ---------------------------------------------------------------------------
// Host orchestration
// ---------------------------------------------------------------------------
extern "C" void kernel_launch(void* const* d_in, const int* in_sizes, int n_in,
                              void* d_out, int out_size, void* d_ws, size_t ws_size,
                              hipStream_t stream) {
    const int N = NPTS, B = BATCH;
    const float* inputs = (const float*)d_in[0];
    const float* W[8]; const float* gv[8]; const float* bv[8];
    for (int i = 1; i <= 7; ++i) {
        W[i]  = (const float*)d_in[1 + (i - 1) * 3];
        gv[i] = (const float*)d_in[2 + (i - 1) * 3];
        bv[i] = (const float*)d_in[3 + (i - 1) * 3];
    }
    const float* L1  = (const float*)d_in[22];
    const float* g8  = (const float*)d_in[23];
    const float* b8  = (const float*)d_in[24];
    const float* L2  = (const float*)d_in[25];
    const float* bl2 = (const float*)d_in[26];
    const float* g9  = (const float*)d_in[27];
    const float* b9  = (const float*)d_in[28];
    const float* L3  = (const float*)d_in[29];
    const float* bl3 = (const float*)d_in[30];
    const float* g10 = (const float*)d_in[31];
    const float* b10 = (const float*)d_in[32];
    const float* L4  = (const float*)d_in[33];
    const float* bl4 = (const float*)d_in[34];

    // deterministic workspace layout (bytes)
    char* ws = (char*)d_ws;
    size_t off = 0;
    auto take = [&](size_t bytes) { char* p = ws + off; off = (off + bytes + 255) & ~(size_t)255; return p; };
    float* xbuf   = (float*)take((size_t)B * 17 * N * 4);
    int*   idx0   = (int*)  take((size_t)B * N * 3 * 4);
    int*   knnidx = (int*)  take((size_t)B * N * 20 * 4);
    float* xc     = (float*)take((size_t)B * 1024 * N * 4);
    float* ubuf   = (float*)take((size_t)B * 256 * N * 4);
    float* vbuf   = (float*)take((size_t)B * 256 * N * 4);
    float* inner  = (float*)take((size_t)B * N * N * 4);   // also reused as conv7 y
    float* xxb    = (float*)take((size_t)B * N * 4);
    float* ssum   = (float*)take(1024 * 4);
    float* ssq    = (float*)take(1024 * 4);
    float* sbuf   = (float*)take(1024 * 4);
    float* tbuf   = (float*)take(1024 * 4);
    float* hbuf   = (float*)take((size_t)B * 2048 * 4);
    float* y7     = inner;  // conv7 output aliases KNN distance buffer

    k_prep<<<(B * 20 * N + 255) / 256, 256, 0, stream>>>(inputs, xbuf, idx0);

    const long xcStride = (long)1024 * N;
    auto edge_block = [&](const float* Wp, const float* gp, const float* bp,
                          const float* Xg, long strideX, const int* idxp, int k,
                          int C, int O, int coffOut) {
        dim3 gg(N / 128, O / 64, B);
        k_gemm<0><<<gg, 256, 0, stream>>>(Wp, 2 * C, 0, Xg, strideX, ubuf, O, N, C, 0);
        k_gemm<1><<<gg, 256, 0, stream>>>(Wp, 2 * C, C, Xg, strideX, vbuf, O, N, C, 0);
        k_zero<<<8, 256, 0, stream>>>(ssum, 1024);
        k_zero<<<8, 256, 0, stream>>>(ssq, 1024);
        k_edge_stats<<<B * N / 16, O, 0, stream>>>(ubuf, vbuf, idxp, k, O, N, 16, ssum, ssq);
        k_finalize<<<(O + 255) / 256, 256, 0, stream>>>(ssum, ssq, gp, bp,
                                                        (float)((long)B * N * k), sbuf, tbuf, O);
        k_edge_out<<<B * N / 16, O, 0, stream>>>(ubuf, vbuf, idxp, k, O, N, 16, sbuf, tbuf,
                                                 xc + (size_t)coffOut * N, xcStride);
    };

    auto knn = [&](const float* Xg, long strideX, int C) {
        k_xx<<<(B * N + 255) / 256, 256, 0, stream>>>(Xg, strideX, C, N, xxb);
        dim3 gg(N / 128, N / 64, B);                       // Gram matrix: inner = X^T X
        k_gemm<2><<<gg, 256, 0, stream>>>(Xg, 0, 0, Xg, strideX, inner, N, N, C, strideX);
        k_topk<<<(B * N + 255) / 256, 256, 0, stream>>>(inner, xxb, knnidx, N);
    };

    // blocks 1-3: fixed 3-NN from inputs
    edge_block(W[1], gv[1], bv[1], xbuf,                 (long)17 * N, idx0, 3, 17,  64,  0);
    edge_block(W[2], gv[2], bv[2], xc,                   xcStride,     idx0, 3, 64,  64,  64);
    edge_block(W[3], gv[3], bv[3], xc + 64 * (size_t)N,  xcStride,     idx0, 3, 64, 128, 128);
    // blocks 4-6: dynamic 20-NN
    knn(xc + 128 * (size_t)N, xcStride, 128);
    edge_block(W[4], gv[4], bv[4], xc + 128 * (size_t)N, xcStride, knnidx, 20, 128, 256, 256);
    knn(xc + 256 * (size_t)N, xcStride, 256);
    edge_block(W[5], gv[5], bv[5], xc + 256 * (size_t)N, xcStride, knnidx, 20, 256, 256, 512);
    knn(xc + 512 * (size_t)N, xcStride, 256);
    edge_block(W[6], gv[6], bv[6], xc + 512 * (size_t)N, xcStride, knnidx, 20, 256, 256, 768);

    // conv7: 1024x1024 pointwise over concat features
    {
        dim3 g7(N / 128, 1024 / 64, B);
        k_gemm<0><<<g7, 256, 0, stream>>>(W[7], 1024, 0, xc, xcStride, y7, 1024, N, 1024, 0);
        k_conv7_stats<<<1024, 256, 0, stream>>>(y7, N, ssum, ssq);
        k_finalize<<<4, 256, 0, stream>>>(ssum, ssq, gv[7], bv[7], (float)((long)B * N),
                                          sbuf, tbuf, 1024);
        k_conv7_out<<<B * 1024, 256, 0, stream>>>(y7, N, sbuf, tbuf, hbuf);
    }

    k_mlp<<<1, 256, 0, stream>>>(hbuf, L1, g8, b8, L2, bl2, g9, b9,
                                 L3, bl3, g10, b10, L4, bl4, (float*)d_out);
}
